// Attention_41403484734016
// MI455X (gfx1250) — matrix operand order repaired and checked by reading, not compiled, not run on hardware
//
#include <hip/hip_runtime.h>
#include <stdint.h>

// ---------- vector types ----------
typedef __attribute__((ext_vector_type(16))) __bf16 v16bf;
typedef __attribute__((ext_vector_type(8)))  __bf16 v8bf;
typedef __attribute__((ext_vector_type(4)))  __bf16 v4bf;
typedef __attribute__((ext_vector_type(8)))  float  v8f;
typedef __attribute__((ext_vector_type(4)))  float  v4f;

union FragU { v16bf v; v8bf h[2]; };

// A-matrix fragment (16x32 bf16): lane&15 = row, K chunks at (lane>>4)*8 and +16
__device__ __forceinline__ v16bf load_fragA(const __bf16* base, int stride, int lane) {
    const __bf16* p = base + (size_t)(lane & 15) * stride + ((lane >> 4) << 3);
    FragU u;
    u.h[0] = *(const v8bf*)(p);
    u.h[1] = *(const v8bf*)(p + 16);
    return u.v;
}

// B-matrix fragment (32x16 bf16) loaded from B^T (N x K row-major):
// lane&15 = column (row of B^T), K chunks at (lane>>4)*8 and +16
__device__ __forceinline__ v16bf load_fragB(const __bf16* base, int stride, int lane) {
    const __bf16* p = base + (size_t)(lane & 15) * stride + ((lane >> 4) << 3);
    FragU u;
    u.h[0] = *(const v8bf*)(p);
    u.h[1] = *(const v8bf*)(p + 16);
    return u.v;
}

__device__ __forceinline__ v8f wmma_bf16(v16bf a, v16bf b, v8f c) {
    return __builtin_amdgcn_wmma_f32_16x16x32_bf16(false, a, false, b, (short)0, c,
                                                   false, false);
}

// ---------- fp32 -> bf16 convert ----------
__global__ void cvt_bf16_kernel(const float* __restrict__ in, __bf16* __restrict__ out, int n) {
    int i = (blockIdx.x * blockDim.x + threadIdx.x) * 4;
    if (i < n) {
        v4f f = *(const v4f*)(in + i);
        v4bf o;
        o[0] = (__bf16)f[0]; o[1] = (__bf16)f[1];
        o[2] = (__bf16)f[2]; o[3] = (__bf16)f[3];
        *(v4bf*)(out + i) = o;
    }
}

// ---------- W (KxN fp32) -> W^T (NxK bf16), LDS-tiled ----------
__global__ void transpose_bf16_kernel(const float* __restrict__ W, __bf16* __restrict__ WT,
                                      int K, int N) {
    __shared__ float tile[32][33];
    int x = blockIdx.x * 32 + threadIdx.x;        // n
    int y0 = blockIdx.y * 32 + threadIdx.y;       // k
#pragma unroll
    for (int j = 0; j < 4; ++j)
        tile[threadIdx.y + j * 8][threadIdx.x] = W[(size_t)(y0 + j * 8) * N + x];
    __syncthreads();
    int xo = blockIdx.y * 32 + threadIdx.x;       // k
    int yo = blockIdx.x * 32 + threadIdx.y;       // n
#pragma unroll
    for (int j = 0; j < 4; ++j)
        WT[(size_t)(yo + j * 8) * K + xo] = (__bf16)tile[threadIdx.x][threadIdx.y + j * 8];
}

// ---------- GEMM: C[MxN] = A[MxK] * B, with B supplied as B^T [NxK], bf16 in, WMMA ----
// block = 256 threads (8 waves), block tile 256(M) x 128(N), wave tile 64x64
// 16 WMMAs per 16 b128 loads per wave per K-step -> 1 load per WMMA.
template <bool F32OUT>
__global__ __launch_bounds__(256) void gemm_bt_kernel(const __bf16* __restrict__ A,
                                                      const __bf16* __restrict__ BT,
                                                      void* __restrict__ C,
                                                      int M, int N, int K) {
    int lane = threadIdx.x & 31;
    int w = threadIdx.x >> 5;
    int m0 = blockIdx.y * 256 + (w >> 1) * 64;
    int n0 = blockIdx.x * 128 + (w & 1) * 64;

    v8f acc[4][4] = {};
    const __bf16* ap[4];
    const __bf16* bp[4];
#pragma unroll
    for (int i = 0; i < 4; ++i) {
        ap[i] = A + (size_t)(m0 + i * 16) * K;
        bp[i] = BT + (size_t)(n0 + i * 16) * K;
    }
    // per-lane base rows for prefetching ahead along K
    const __bf16* apf = ap[0] + (size_t)(lane & 15) * K;
    const __bf16* bpf = bp[0] + (size_t)(lane & 15) * K;

    for (int k0 = 0; k0 < K; k0 += 32) {
        // prefetch ~4 iterations ahead into cache (speculative, OOB dropped)
        __builtin_prefetch(apf + k0 + 128, 0, 1);
        __builtin_prefetch(bpf + k0 + 128, 0, 1);
        v16bf a[4], b[4];
#pragma unroll
        for (int i = 0; i < 4; ++i) a[i] = load_fragA(ap[i] + k0, K, lane);
#pragma unroll
        for (int j = 0; j < 4; ++j) b[j] = load_fragB(bp[j] + k0, K, lane);
#pragma unroll
        for (int i = 0; i < 4; ++i)
#pragma unroll
            for (int j = 0; j < 4; ++j)
                acc[i][j] = wmma_bf16(a[i], b[j], acc[i][j]);
    }

    int r0 = (lane >> 4) << 3;   // row base within 16-row tile
    int cc = lane & 15;          // column within 16-col tile
#pragma unroll
    for (int i = 0; i < 4; ++i) {
#pragma unroll
        for (int r = 0; r < 8; ++r) {
            size_t rowoff = (size_t)(m0 + i * 16 + r0 + r) * N + n0 + cc;
#pragma unroll
            for (int j = 0; j < 4; ++j) {
                if (F32OUT) {
                    ((float*)C)[rowoff + j * 16] = acc[i][j][r];
                } else {
                    ((__bf16*)C)[rowoff + j * 16] = (__bf16)acc[i][j][r];
                }
            }
        }
    }
}

// ---------- RoPE in-place on packed QKV (rows = b*S+s, 4096 cols = [Q|K|V]) -------
__global__ void rope_kernel(__bf16* __restrict__ QKV,
                            const float* __restrict__ fcos,
                            const float* __restrict__ fsin) {
    unsigned tid = blockIdx.x * blockDim.x + threadIdx.x;   // B*S*24*64 threads
    int d2 = tid & 63;
    int hh = (tid >> 6) % 24;
    unsigned bs = tid / (64u * 24u);
    int s = bs & 2047;
    int col = (hh < 16) ? (hh * 128 + 2 * d2) : (2048 + (hh - 16) * 128 + 2 * d2);
    __bf16* p = QKV + (size_t)bs * 4096 + col;
    float tr = (float)p[0];
    float ti = (float)p[1];
    float c = fcos[s * 64 + d2];
    float sn = fsin[s * 64 + d2];
    p[0] = (__bf16)(tr * c - ti * sn);
    p[1] = (__bf16)(tr * sn + ti * c);
}

// ---------- V scatter: QKV V-cols -> Vt[((b*8+kh)*128+d)*2048 + s] ----------
__global__ void vt_kernel(const __bf16* __restrict__ QKV, __bf16* __restrict__ Vt) {
    unsigned tid = blockIdx.x * blockDim.x + threadIdx.x;   // 2*8*128*2048 = 2^22
    int s = tid & 2047;
    int d = (tid >> 11) & 127;
    int kh = (tid >> 18) & 7;
    int b = tid >> 21;
    Vt[tid] = QKV[((size_t)(b * 2048 + s)) * 4096 + 3072 + kh * 128 + d];
}

// ---------- Flash attention: 1 wave per (b, h, 16-row q tile) ----------
// V tile for each 32-key block is staged into LDS with GLOBAL_LOAD_ASYNC_TO_LDS_B128
// (ASYNCcnt), overlapping the QK^T WMMAs and softmax; P.V reads V from LDS.
__global__ __launch_bounds__(32) void attn_kernel(const __bf16* __restrict__ QKV,
                                                  const __bf16* __restrict__ Vt,
                                                  __bf16* __restrict__ Ob) {
    __shared__ __align__(16) __bf16 lds_p[16 * 32];
    __shared__ __align__(16) __bf16 lds_o[16 * 128];
    __shared__ __align__(16) __bf16 vstage[128 * 32];   // [d][key] tile, 8KB

    int lane = threadIdx.x;
    int bid = blockIdx.x;
    int qt = bid & 127;
    int h = (bid >> 7) & 15;
    int b = bid >> 11;
    int kh = h >> 1;
    int q0 = qt * 16;
    const float scl = 0.08838834764831845f; // 1/sqrt(128)

    // Q fragments for this 16x128 tile (A-matrix layout), reused across K blocks
    v16bf qf[4];
    const __bf16* qbase = QKV + ((size_t)(b * 2048 + q0)) * 4096 + h * 128;
#pragma unroll
    for (int kk = 0; kk < 4; ++kk) qf[kk] = load_fragA(qbase + kk * 32, 4096, lane);

    float m[8], l[8];
    v8f o[8] = {};
#pragma unroll
    for (int r = 0; r < 8; ++r) { m[r] = -1e30f; l[r] = 0.0f; }

    int rowbase = (lane >> 4) << 3;
    int cl = lane & 15;
    int nkb = (q0 + 16 + 31) >> 5;   // 32-key blocks needed under causal mask
    const __bf16* kbase = QKV + ((size_t)b * 2048) * 4096 + 2048 + kh * 128;
    const __bf16* vbase = Vt + ((size_t)((b * 8 + kh) * 128)) * 2048;

    // per-lane LDS scatter offsets for the async V-tile copy (16 chunks of 16B)
    const uint32_t vstage_base = (uint32_t)(uintptr_t)(&vstage[0]);

    for (int kb = 0; kb < nkb; ++kb) {
        // ---- issue async copy of V tile (128 d-rows x 32 keys) into LDS ----
        {
            const __bf16* src = vbase + kb * 32;
#pragma unroll
            for (int i = 0; i < 16; ++i) {
                int chunk = lane + 32 * i;          // 512 chunks of 8 bf16 (16B)
                int d = chunk >> 2;
                int c = (chunk & 3) * 8;
                uint64_t g = (uint64_t)(uintptr_t)(src + (size_t)d * 2048 + c);
                uint32_t ldsoff = vstage_base + (uint32_t)(d * 32 + c) * 2u;
                asm volatile("global_load_async_to_lds_b128 %0, %1, off"
                             :: "v"(ldsoff), "v"(g) : "memory");
            }
        }

        // ---- S = Q K^T (two 16x16 column tiles covering 32 keys) ----
        v8f s0 = {}, s1 = {};
        const __bf16* kp = kbase + (size_t)(kb * 32) * 4096;
#pragma unroll
        for (int kk = 0; kk < 4; ++kk) {
            v16bf k0 = load_fragB(kp + kk * 32, 4096, lane);
            v16bf k1 = load_fragB(kp + (size_t)16 * 4096 + kk * 32, 4096, lane);
            s0 = wmma_bf16(qf[kk], k0, s0);
            s1 = wmma_bf16(qf[kk], k1, s1);
        }
        // scale + causal mask + block row-max
        int key0 = kb * 32 + cl;
        int key1 = key0 + 16;
        float bm[8];
#pragma unroll
        for (int r = 0; r < 8; ++r) {
            int q = q0 + rowbase + r;
            float v0 = s0[r] * scl;
            float v1 = s1[r] * scl;
            if (key0 > q) v0 = -1e30f;
            if (key1 > q) v1 = -1e30f;
            s0[r] = v0; s1[r] = v1;
            bm[r] = fmaxf(v0, v1);
        }
#pragma unroll
        for (int r = 0; r < 8; ++r) {
            float v = bm[r];
            v = fmaxf(v, __shfl_xor(v, 1));
            v = fmaxf(v, __shfl_xor(v, 2));
            v = fmaxf(v, __shfl_xor(v, 4));
            v = fmaxf(v, __shfl_xor(v, 8));
            bm[r] = v;
        }
        // online softmax update; P -> LDS (C-layout to row-major 16x32)
        float alpha[8], ps[8];
#pragma unroll
        for (int r = 0; r < 8; ++r) {
            float mn = fmaxf(m[r], bm[r]);
            alpha[r] = __expf(m[r] - mn);
            m[r] = mn;
            float p0 = __expf(s0[r] - mn);
            float p1 = __expf(s1[r] - mn);
            int row = rowbase + r;
            lds_p[row * 32 + cl]      = (__bf16)p0;
            lds_p[row * 32 + 16 + cl] = (__bf16)p1;
            ps[r] = p0 + p1;
        }
#pragma unroll
        for (int r = 0; r < 8; ++r) {
            float v = ps[r];
            v += __shfl_xor(v, 1);
            v += __shfl_xor(v, 2);
            v += __shfl_xor(v, 4);
            v += __shfl_xor(v, 8);
            l[r] = l[r] * alpha[r] + v;
        }
#pragma unroll
        for (int t = 0; t < 8; ++t)
#pragma unroll
            for (int r = 0; r < 8; ++r) o[t][r] *= alpha[r];

        __syncthreads();                              // P writes visible (dscnt fence)
        asm volatile("s_wait_asynccnt 0x0" ::: "memory");  // V tile landed in LDS

        v16bf pf = load_fragA(lds_p, 32, lane);       // P as A-matrix (16x32)
#pragma unroll
        for (int t = 0; t < 8; ++t) {
            v16bf vf = load_fragB(&vstage[t * 16 * 32], 32, lane);  // from LDS
            o[t] = wmma_bf16(pf, vf, o[t]);
        }
        __syncthreads();   // ds reads done before next block's async writes / P writes
    }

    // finalize with softmax-sink denominator: l + exp(-m)
    float inv[8];
#pragma unroll
    for (int r = 0; r < 8; ++r) inv[r] = 1.0f / (l[r] + __expf(-m[r]));
#pragma unroll
    for (int t = 0; t < 8; ++t)
#pragma unroll
        for (int r = 0; r < 8; ++r) {
            int row = rowbase + r;
            lds_o[row * 128 + t * 16 + cl] = (__bf16)(o[t][r] * inv[r]);
        }
    __syncthreads();
    // coalesced 16B stores of the 16x128 tile
#pragma unroll
    for (int i = 0; i < 8; ++i) {
        int chunk = lane + 32 * i;
        int row = chunk >> 4;
        int c8 = (chunk & 15) * 8;
        v8bf val = *(const v8bf*)(lds_o + row * 128 + c8);
        *(v8bf*)(Ob + ((size_t)(b * 2048 + q0 + row)) * 2048 + h * 128 + c8) = val;
    }
}

// ---------- host-side orchestration ----------
extern "C" void kernel_launch(void* const* d_in, const int* in_sizes, int n_in,
                              void* d_out, int out_size, void* d_ws, size_t ws_size,
                              hipStream_t stream) {
    (void)in_sizes; (void)n_in; (void)out_size; (void)ws_size;
    const float* x  = (const float*)d_in[0];
    const float* fc = (const float*)d_in[1];
    const float* fs = (const float*)d_in[2];
    const float* wq = (const float*)d_in[3];
    const float* wk = (const float*)d_in[4];
    const float* wv = (const float*)d_in[5];
    const float* wo = (const float*)d_in[6];
    float* out = (float*)d_out;

    // workspace layout (bf16 elements)
    __bf16* xb    = (__bf16*)d_ws;                          //  8M  (4096 x 2048)
    __bf16* wqkvT = xb    + (size_t)4096 * 2048;            //  8M  (4096 x 2048)
    __bf16* woT   = wqkvT + (size_t)4096 * 2048;            //  4M  (2048 x 2048)
    __bf16* QKV   = woT   + (size_t)2048 * 2048;            // 16M  (4096 x 4096)
    __bf16* Vt    = QKV   + (size_t)4096 * 4096;            //  4M  (2*8*128 x 2048)
    __bf16* Ob    = Vt    + (size_t)2 * 8 * 128 * 2048;     //  8M  (4096 x 2048)

    dim3 tb(32, 8);
    cvt_bf16_kernel<<<8192, 256, 0, stream>>>(x, xb, 4096 * 2048);
    transpose_bf16_kernel<<<dim3(64, 64), tb, 0, stream>>>(wq, wqkvT, 2048, 2048);
    transpose_bf16_kernel<<<dim3(32, 64), tb, 0, stream>>>(wk, wqkvT + (size_t)2048 * 2048, 2048, 1024);
    transpose_bf16_kernel<<<dim3(32, 64), tb, 0, stream>>>(wv, wqkvT + (size_t)3072 * 2048, 2048, 1024);
    transpose_bf16_kernel<<<dim3(64, 64), tb, 0, stream>>>(wo, woT, 2048, 2048);

    // fused QKV projection: [x] @ [wq|wk|wv]   (block tile 256x128)
    gemm_bt_kernel<false><<<dim3(32, 16), 256, 0, stream>>>(xb, wqkvT, QKV, 4096, 4096, 2048);

    rope_kernel<<<24576, 256, 0, stream>>>(QKV, fc, fs);   // 2*2048*24*64 threads
    vt_kernel<<<16384, 256, 0, stream>>>(QKV, Vt);         // 2^22 threads

    attn_kernel<<<4096, 32, 0, stream>>>(QKV, Vt, Ob);     // (b,h,qtile) = 2*16*128

    // output projection -> fp32 d_out
    gemm_bt_kernel<true><<<dim3(16, 16), 256, 0, stream>>>(Ob, woT, out, 4096, 2048, 2048);
}
